// MultiheadSelfAttention_10651518894457
// MI455X (gfx1250) — matrix operand order repaired and checked by reading, not compiled, not run on hardware
//
#include <hip/hip_runtime.h>
#include <hip/hip_bf16.h>
#include <math.h>

#define Bn 4
#define Sn 1025
#define En 1024
#define Hn 16
#define DH 64
#define Tn 1024   /* S-1 tokens that attend */
#define EPSf 1e-5f

typedef __attribute__((ext_vector_type(16))) __bf16 v16bf;
typedef __attribute__((ext_vector_type(8)))  __bf16 v8bf;
typedef __attribute__((ext_vector_type(8)))  float  v8f;
typedef __attribute__((ext_vector_type(4))) unsigned int u32x4;
typedef __attribute__((ext_vector_type(8))) int i32x8;
typedef __attribute__((ext_vector_type(4))) int i32x4;

// ---------------------------------------------------------------------------
// WMMA helpers (CDNA5 wave32, D = A(16x32 bf16) * B(32x16 bf16) + C(16x16 f32))
// ---------------------------------------------------------------------------
static __device__ __forceinline__ v8f wmma_bf16(v16bf a, v16bf b, v8f c) {
  return __builtin_amdgcn_wmma_f32_16x16x32_bf16(false, a, false, b, (short)0, c,
                                                 false, false);
}

// A-operand: lanes 0-15 hold row M=lane, K = k0+0..7 (elts 0..7) and k0+16..23
// (elts 8..15); lanes 16-31 hold K = k0+8..15 and k0+24..31.  Two 16B loads.
static __device__ __forceinline__ v16bf load_a16(const __bf16* base, int stride,
                                                 int row, int k0, bool hi) {
  const __bf16* p = base + (size_t)row * stride + k0 + (hi ? 8 : 0);
  v8bf lo = *(const v8bf*)p;
  v8bf h2 = *(const v8bf*)(p + 16);
  v16bf a;
#pragma unroll
  for (int i = 0; i < 8; ++i) { a[i] = lo[i]; a[i + 8] = h2[i]; }
  return a;
}

// B-operand B[k][n]: lane n(0-15) holds column n, K=k0+0..7 and k0+16..23; lanes
// 16-31 hold column n-16, K=k0+8..15 and k0+24..31.  Column n is a contiguous row of `base`.
static __device__ __forceinline__ v16bf load_b16(const __bf16* base, int stride,
                                                 int col, int k0, bool hi) {
  const __bf16* p = base + (size_t)col * stride + k0 + (hi ? 8 : 0);
  v8bf lo = *(const v8bf*)p;
  v8bf h2 = *(const v8bf*)(p + 16);
  v16bf b;
#pragma unroll
  for (int i = 0; i < 8; ++i) { b[i] = lo[i]; b[i + 8] = h2[i]; }
  return b;
}

// ---------------------------------------------------------------------------
// Tensor Data Mover: stage a (32 elem x 256 row) bf16 tile of W into LDS.
// D# per CDNA5 ISA:  data_size=2B, tile_dim0=32 (64B rows),
// tile_dim1=256, tensor_dim0_stride=1024, LDS pad = 4 dwords per 16 dwords
// (row pitch 80B -> conflict-free ds_load_b128 across lanes).
// This toolchain uses the 6-arg builtin (g0, g1, g2, g3, g4, cpol).
// ---------------------------------------------------------------------------
static __device__ __forceinline__ void tdm_load_tile(unsigned lds_addr,
                                                     const __bf16* gptr) {
  unsigned long long ga = (unsigned long long)(uintptr_t)gptr;
  u32x4 g0;
  g0[0] = 1u;                                  // count=1, user descriptor
  g0[1] = lds_addr;                            // LDS byte address
  g0[2] = (unsigned)ga;                        // global_addr[31:0]
  g0[3] = (unsigned)(ga >> 32) | 0x80000000u;  // global_addr[56:32] | type=2
  i32x8 g1;
  g1[0] = 0x06D10000;        // data_size=2B | pad_en | pad_interval=16dw | pad=4dw
  g1[1] = (int)(32u << 16);  // tensor_dim0 = 32 (bits 79:48, low half)
  g1[2] = (int)(256u << 16); // tensor_dim0 hi | tensor_dim1 = 256
  g1[3] = (int)(32u << 16);  // tensor_dim1 hi | tile_dim0 = 32
  g1[4] = 256;               // tile_dim1 = 256, tile_dim2 = 0
  g1[5] = 1024;              // tensor_dim0_stride = 1024 (data_size units)
  g1[6] = 0;
  g1[7] = 0;
  i32x4 z4 = {0, 0, 0, 0};
  i32x8 z8 = {0, 0, 0, 0, 0, 0, 0, 0};
  __builtin_amdgcn_tensor_load_to_lds(g0, g1, z4, z4, z8, 0);
}

// ---------------------------------------------------------------------------
// fp32 -> bf16 convert
// ---------------------------------------------------------------------------
__global__ void k_cvt_bf16(const float* __restrict__ src, __bf16* __restrict__ dst,
                           int n) {
  int i = blockIdx.x * blockDim.x + threadIdx.x;
  if (i < n) dst[i] = (__bf16)src[i];
}

// ---------------------------------------------------------------------------
// C = A(MxE) @ W^T(ExE) + bias.  Block = 4 waves, 16 rows x 256 cols.
// W tiles are TDM-staged into LDS (double buffered, TENSORcnt-tracked);
// one fetch per block serves all 4 waves.  A streams from global.
// ---------------------------------------------------------------------------
#define LDS_PITCH 40  /* bf16 elems per LDS row: 32 data + 8 TDM pad */

__global__ void k_gemm_bt(const __bf16* __restrict__ A0, const __bf16* __restrict__ W,
                          const float* __restrict__ bias, __bf16* __restrict__ outb0,
                          float* __restrict__ outf0, int M, size_t Astride,
                          size_t Ostride) {
  __shared__ __bf16 smem[2][256 * LDS_PITCH];
  const __bf16* A = A0 + (size_t)blockIdx.z * Astride;
  int lane = threadIdx.x, l16 = lane & 15;
  bool hi = lane >= 16;
  int wave = threadIdx.y;
  int m0 = blockIdx.x * 16;
  int nblk = blockIdx.y * 256;               // block's 256 output cols
  int row = m0 + l16; if (row >= M) row = M - 1;

  unsigned lds0 = (unsigned)(uintptr_t)(&smem[0][0]);
  unsigned lds1 = (unsigned)(uintptr_t)(&smem[1][0]);
  if (wave == 0) {
    tdm_load_tile(lds0, W + (size_t)nblk * En + 0);
    tdm_load_tile(lds1, W + (size_t)nblk * En + 32);
  }

  v8f acc[4] = {};
  for (int step = 0; step < 32; ++step) {
    int k0 = step * 32;
    if (wave == 0) {
      if (step < 31) __builtin_amdgcn_s_wait_tensorcnt((short)1);
      else           __builtin_amdgcn_s_wait_tensorcnt((short)0);
    }
    __syncthreads();
    v16bf a = load_a16(A, En, row, k0, hi);
    if (k0 + 64 < En)
      __builtin_prefetch(A + (size_t)row * En + k0 + 64, 0, 0);
    const __bf16* sb = smem[step & 1];
#pragma unroll
    for (int t = 0; t < 4; ++t) {
      int lrow = wave * 64 + 16 * t + l16;   // tile row = output col - nblk
      const __bf16* p = sb + lrow * LDS_PITCH + (hi ? 8 : 0);
      v8bf lo = *(const v8bf*)p;
      v8bf h2 = *(const v8bf*)(p + 16);
      v16bf b;
#pragma unroll
      for (int i = 0; i < 8; ++i) { b[i] = lo[i]; b[i + 8] = h2[i]; }
      acc[t] = wmma_bf16(a, b, acc[t]);
    }
    __syncthreads();                          // readers done before overwrite
    if (wave == 0 && step + 2 < 32)
      tdm_load_tile((step & 1) ? lds1 : lds0,
                    W + (size_t)nblk * En + (size_t)(step + 2) * 32);
  }

  int mbase = m0 + (hi ? 8 : 0);
#pragma unroll
  for (int t = 0; t < 4; ++t) {
    int col = nblk + wave * 64 + 16 * t + l16;
    float bv = bias ? bias[col] : 0.f;
#pragma unroll
    for (int r = 0; r < 8; ++r) {
      int rr = mbase + r;
      if (rr < M) {
        float v = acc[t][r] + bv;
        if (outb0) (outb0 + (size_t)blockIdx.z * Ostride)[(size_t)rr * En + col] = (__bf16)v;
        else       (outf0 + (size_t)blockIdx.z * Ostride)[(size_t)rr * En + col] = v;
      }
    }
  }
}

// ---------------------------------------------------------------------------
// Build v^T and cproj^T : [bh][d][token]
// ---------------------------------------------------------------------------
__global__ void k_build_T(const __bf16* __restrict__ vbf, const __bf16* __restrict__ cpbf,
                          __bf16* __restrict__ vT, __bf16* __restrict__ cpT) {
  int i = blockIdx.x * blockDim.x + threadIdx.x;  // B*H*DH*Tn
  if (i >= Bn * Hn * DH * Tn) return;
  int n = i & 1023, d = (i >> 10) & 63, h = (i >> 16) & 15, b = i >> 20;
  size_t src = ((size_t)(b * Sn + 1 + n)) * En + h * DH + d;
  vT[i] = vbf[src];
  cpT[i] = cpbf[src];
}

// ---------------------------------------------------------------------------
// per-(b,h,token): q_norm2 (clipped), cq = c.q, ck = c.k
// ---------------------------------------------------------------------------
__global__ void k_norms(const __bf16* __restrict__ qbf, const __bf16* __restrict__ kbf,
                        float* __restrict__ qn2, float* __restrict__ cqv,
                        float* __restrict__ ckv) {
  int i = blockIdx.x * blockDim.x + threadIdx.x;  // B*H*Tn
  if (i >= Bn * Hn * Tn) return;
  int n = i & 1023, h = (i >> 10) & 15, b = i >> 14;
  const __bf16* qr = qbf + ((size_t)(b * Sn + 1 + n)) * En + h * DH;
  const __bf16* kr = kbf + ((size_t)(b * Sn + 1 + n)) * En + h * DH;
  const __bf16* cr = qbf + ((size_t)(b * Sn)) * En + h * DH;
  float sq = 0.f, scq = 0.f, sck = 0.f;
  for (int d = 0; d < DH; ++d) {
    float qv = (float)qr[d], kv = (float)kr[d], cv = (float)cr[d];
    sq += qv * qv; scq += cv * qv; sck += cv * kv;
  }
  qn2[i] = fmaxf(sq, EPSf);
  cqv[i] = scq;
  ckv[i] = sck;
}

__global__ void k_cnorm(const __bf16* __restrict__ qbf, float* __restrict__ cn2) {
  int i = blockIdx.x * blockDim.x + threadIdx.x;  // B*H
  if (i >= Bn * Hn) return;
  int h = i & 15, b = i >> 4;
  const __bf16* cr = qbf + ((size_t)(b * Sn)) * En + h * DH;
  float s = 0.f;
  for (int d = 0; d < DH; ++d) { float c = (float)cr[d]; s += c * c; }
  cn2[i] = fmaxf(s, EPSf);
}

// ---------------------------------------------------------------------------
// raw logits = (q.k)/sqrt(64*qn2) -> written straight into d_out attn region
// ---------------------------------------------------------------------------
__global__ void k_logits(const __bf16* __restrict__ qbf, const __bf16* __restrict__ kbf,
                         const float* __restrict__ qn2, float* __restrict__ attn) {
  int lane = threadIdx.x, l16 = lane & 15;
  bool hi = lane >= 16;
  int q0 = blockIdx.x * 16;
  int kg = blockIdx.y * blockDim.y + threadIdx.y;
  int bh = blockIdx.z, b = bh >> 4, h = bh & 15;
  const __bf16* qbase = qbf + ((size_t)(b * Sn + 1)) * En + h * DH;
  const __bf16* kbase = kbf + ((size_t)(b * Sn + 1)) * En + h * DH;
  int n0 = kg * 64;
  v8f acc[4] = {};
#pragma unroll
  for (int k0 = 0; k0 < DH; k0 += 32) {
    v16bf a = load_a16(qbase, En, q0 + l16, k0, hi);
#pragma unroll
    for (int t = 0; t < 4; ++t) {
      v16bf bfr = load_b16(kbase, En, n0 + 16 * t + l16, k0, hi);
      acc[t] = wmma_bf16(a, bfr, acc[t]);
    }
  }
  int mbase = q0 + (hi ? 8 : 0);
  float* abh = attn + (size_t)bh * Tn * Tn;
#pragma unroll
  for (int t = 0; t < 4; ++t) {
    int col = n0 + 16 * t + l16;
#pragma unroll
    for (int r = 0; r < 8; ++r) {
      int q = mbase + r;
      float s = rsqrtf(64.f * qn2[(bh << 10) + q]);
      abh[(size_t)q * Tn + col] = acc[t][r] * s;
    }
  }
}

// ---------------------------------------------------------------------------
// per-row softmax stats for both variants
// ---------------------------------------------------------------------------
__global__ void k_stats(const float* __restrict__ attn, const float* __restrict__ qn2,
                        const float* __restrict__ cqv, const float* __restrict__ cn2,
                        float* __restrict__ stats) {
  int lane = threadIdx.x;
  int rowid = blockIdx.x * blockDim.y + threadIdx.y;
  int bh = rowid >> 10;
  const float* lr = attn + (size_t)rowid * Tn;
  const float* cqr = cqv + (bh << 10);
  float invs = rsqrtf(qn2[rowid] * cn2[bh]);
  float l[32];
  float m1 = -1e30f, m2 = -1e30f;
#pragma unroll
  for (int i = 0; i < 32; ++i) {
    float x = lr[lane + 32 * i];
    l[i] = x;
    m1 = fmaxf(m1, x);
    m2 = fmaxf(m2, x * cqr[lane + 32 * i] * invs);
  }
  for (int o = 16; o; o >>= 1) {
    m1 = fmaxf(m1, __shfl_xor(m1, o, 32));
    m2 = fmaxf(m2, __shfl_xor(m2, o, 32));
  }
  float s1 = 0.f, s2 = 0.f;
#pragma unroll
  for (int i = 0; i < 32; ++i) {
    s1 += __expf(l[i] - m1);
    s2 += __expf(l[i] * cqr[lane + 32 * i] * invs - m2);
  }
  for (int o = 16; o; o >>= 1) {
    s1 += __shfl_xor(s1, o, 32);
    s2 += __shfl_xor(s2, o, 32);
  }
  if (lane == 0) *(float4*)(stats + (size_t)rowid * 4) = make_float4(m1, s1, m2, s2);
}

// ---------------------------------------------------------------------------
// cls row: attn_o2c = softmax(ck/sqrt(64*cn2)); cls_out = attn_o2c @ v
// ---------------------------------------------------------------------------
__global__ void k_o2c(const float* __restrict__ ckv, const float* __restrict__ cn2,
                      const __bf16* __restrict__ vbf, __bf16* __restrict__ valbf) {
  int bh = blockIdx.x, lane = threadIdx.x;
  int b = bh >> 4, h = bh & 15;
  const float* ckr = ckv + (bh << 10);
  float invD = rsqrtf(64.f * cn2[bh]);
  float m = -1e30f;
  for (int i = lane; i < Tn; i += 32) m = fmaxf(m, ckr[i] * invD);
  for (int o = 16; o; o >>= 1) m = fmaxf(m, __shfl_xor(m, o, 32));
  float s = 0.f;
  for (int i = lane; i < Tn; i += 32) s += __expf(ckr[i] * invD - m);
  for (int o = 16; o; o >>= 1) s += __shfl_xor(s, o, 32);
  float inv_s = 1.f / s;
  int d0 = lane * 2, d1 = d0 + 1;
  float a0 = 0.f, a1 = 0.f;
  const __bf16* vb = vbf + ((size_t)(b * Sn + 1)) * En + h * DH;
  for (int k = 0; k < Tn; ++k) {
    float w = __expf(ckr[k] * invD - m) * inv_s;
    a0 += w * (float)vb[(size_t)k * En + d0];
    a1 += w * (float)vb[(size_t)k * En + d1];
  }
  __bf16* dst = valbf + ((size_t)(b * Sn)) * En + h * DH;
  dst[d0] = (__bf16)a0;
  dst[d1] = (__bf16)a1;
}

// ---------------------------------------------------------------------------
// Fused: normalize attn in place (p1), p2 = softmax(l*cq*invs), and
// O = p1@v + p2@cproj via WMMA.  One wave per (bh, 16-q tile).
// ---------------------------------------------------------------------------
__global__ void k_pv(float* __restrict__ attn, const float* __restrict__ stats,
                     const float* __restrict__ qn2, const float* __restrict__ cqv,
                     const float* __restrict__ cn2, const __bf16* __restrict__ vT,
                     const __bf16* __restrict__ cpT, __bf16* __restrict__ valbf) {
  int lane = threadIdx.x, l16 = lane & 15;
  bool hi = lane >= 16;
  int q0 = blockIdx.x * 16;
  int bh = blockIdx.y, b = bh >> 4, h = bh & 15;
  int qrow = q0 + l16;
  int rowid = (bh << 10) + qrow;
  float4 st = *(const float4*)(stats + (size_t)rowid * 4);
  float m1 = st.x, inv_s1 = 1.f / st.y, m2 = st.z, inv_s2 = 1.f / st.w;
  float invs = rsqrtf(qn2[rowid] * cn2[bh]);
  float* lrow = attn + (size_t)rowid * Tn;
  const float* cqr = cqv + (bh << 10);
  const __bf16* vTb = vT + (size_t)bh * DH * Tn;
  const __bf16* cpTb = cpT + (size_t)bh * DH * Tn;
  v8f acc1[4] = {}, acc2[4] = {};
  for (int kc = 0; kc < Tn; kc += 32) {
    int off = kc + (hi ? 8 : 0);
    v8f lo = *(const v8f*)(lrow + off);
    v8f h2 = *(const v8f*)(lrow + off + 16);
    v8f cql = *(const v8f*)(cqr + off);
    v8f cqh = *(const v8f*)(cqr + off + 16);
    v8f p1l, p1h;
    v16bf a1, a2;
#pragma unroll
    for (int i = 0; i < 8; ++i) {
      float pa = __expf(lo[i] - m1) * inv_s1;
      float pb = __expf(h2[i] - m1) * inv_s1;
      p1l[i] = pa; p1h[i] = pb;
      a1[i] = (__bf16)pa; a1[i + 8] = (__bf16)pb;
      float qa = __expf(lo[i] * cql[i] * invs - m2) * inv_s2;
      float qb = __expf(h2[i] * cqh[i] * invs - m2) * inv_s2;
      a2[i] = (__bf16)qa; a2[i + 8] = (__bf16)qb;
    }
    *(v8f*)(lrow + off) = p1l;
    *(v8f*)(lrow + off + 16) = p1h;
#pragma unroll
    for (int t = 0; t < 4; ++t) {
      v16bf bv = load_b16(vTb, Tn, 16 * t + l16, kc, hi);
      acc1[t] = wmma_bf16(a1, bv, acc1[t]);
      v16bf bc2 = load_b16(cpTb, Tn, 16 * t + l16, kc, hi);
      acc2[t] = wmma_bf16(a2, bc2, acc2[t]);
    }
  }
  int mbase = q0 + (hi ? 8 : 0);
#pragma unroll
  for (int t = 0; t < 4; ++t) {
    int d = 16 * t + l16;
#pragma unroll
    for (int r = 0; r < 8; ++r) {
      int tok = mbase + r;
      valbf[((size_t)(b * Sn + 1 + tok)) * En + h * DH + d] =
          (__bf16)(acc1[t][r] + acc2[t][r]);
    }
  }
}

// ---------------------------------------------------------------------------
extern "C" void kernel_launch(void* const* d_in, const int* in_sizes, int n_in,
                              void* d_out, int out_size, void* d_ws, size_t ws_size,
                              hipStream_t stream) {
  (void)in_sizes; (void)n_in; (void)out_size; (void)ws_size;
  const float* x  = (const float*)d_in[0];
  const float* Wq = (const float*)d_in[1];
  const float* bq = (const float*)d_in[2];
  const float* Wk = (const float*)d_in[3];
  const float* bk = (const float*)d_in[4];
  const float* Wv = (const float*)d_in[5];
  const float* bv = (const float*)d_in[6];
  const float* Wo = (const float*)d_in[7];
  const float* bo = (const float*)d_in[8];
  const float* Wc = (const float*)d_in[9];
  const float* bc = (const float*)d_in[10];
  float* out  = (float*)d_out;                          // (B,S,E)
  float* attn = out + (size_t)Bn * Sn * En;             // (B,H,Tn,Tn)

  char* wsb = (char*)d_ws;
  size_t off = 0;
  auto alloc = [&](size_t bytes) -> char* {
    char* p = wsb + off;
    off = (off + bytes + 255) & ~(size_t)255;
    return p;
  };
  const size_t SE = (size_t)Sn * En;
  __bf16* xbf  = (__bf16*)alloc(Bn * SE * 2);
  __bf16* wqb  = (__bf16*)alloc((size_t)En * En * 2);
  __bf16* wkb  = (__bf16*)alloc((size_t)En * En * 2);
  __bf16* wvb  = (__bf16*)alloc((size_t)En * En * 2);
  __bf16* wcb  = (__bf16*)alloc((size_t)En * En * 2);
  __bf16* wob  = (__bf16*)alloc((size_t)En * En * 2);
  __bf16* qbf  = (__bf16*)alloc(Bn * SE * 2);
  __bf16* kbf  = (__bf16*)alloc(Bn * SE * 2);
  __bf16* vbf  = (__bf16*)alloc(Bn * SE * 2);
  __bf16* cpbf = (__bf16*)alloc(Bn * SE * 2);
  __bf16* vT   = (__bf16*)alloc((size_t)Bn * Hn * DH * Tn * 2);
  __bf16* cpT  = (__bf16*)alloc((size_t)Bn * Hn * DH * Tn * 2);
  float*  qn2  = (float*)alloc((size_t)Bn * Hn * Tn * 4);
  float*  cqv  = (float*)alloc((size_t)Bn * Hn * Tn * 4);
  float*  ckv  = (float*)alloc((size_t)Bn * Hn * Tn * 4);
  float*  cn2  = (float*)alloc((size_t)Bn * Hn * 4);
  float*  stats= (float*)alloc((size_t)Bn * Hn * Tn * 16);
  __bf16* valbf= (__bf16*)alloc(Bn * SE * 2);

  int nX = Bn * Sn * En, nW = En * En;
  k_cvt_bf16<<<(nX + 255) / 256, 256, 0, stream>>>(x,  xbf, nX);
  k_cvt_bf16<<<(nW + 255) / 256, 256, 0, stream>>>(Wq, wqb, nW);
  k_cvt_bf16<<<(nW + 255) / 256, 256, 0, stream>>>(Wk, wkb, nW);
  k_cvt_bf16<<<(nW + 255) / 256, 256, 0, stream>>>(Wv, wvb, nW);
  k_cvt_bf16<<<(nW + 255) / 256, 256, 0, stream>>>(Wc, wcb, nW);
  k_cvt_bf16<<<(nW + 255) / 256, 256, 0, stream>>>(Wo, wob, nW);

  // projections (all 1025 rows; q row 0 == cls projection "c")
  dim3 gP((Sn + 15) / 16, 4, Bn), tP(32, 4);
  k_gemm_bt<<<gP, tP, 0, stream>>>(xbf, wqb, bq, qbf,  nullptr, Sn, SE, SE);
  k_gemm_bt<<<gP, tP, 0, stream>>>(xbf, wkb, bk, kbf,  nullptr, Sn, SE, SE);
  k_gemm_bt<<<gP, tP, 0, stream>>>(xbf, wvb, bv, vbf,  nullptr, Sn, SE, SE);
  k_gemm_bt<<<gP, tP, 0, stream>>>(xbf, wcb, bc, cpbf, nullptr, Sn, SE, SE);

  int nT = Bn * Hn * DH * Tn;
  k_build_T<<<(nT + 255) / 256, 256, 0, stream>>>(vbf, cpbf, vT, cpT);

  int nN = Bn * Hn * Tn;
  k_norms<<<(nN + 255) / 256, 256, 0, stream>>>(qbf, kbf, qn2, cqv, ckv);
  k_cnorm<<<1, Bn * Hn, 0, stream>>>(qbf, cn2);

  k_logits<<<dim3(Tn / 16, 4, Bn * Hn), dim3(32, 4), 0, stream>>>(qbf, kbf, qn2, attn);

  k_stats<<<(Bn * Hn * Tn) / 8, dim3(32, 8), 0, stream>>>(attn, qn2, cqv, cn2, stats);

  k_o2c<<<Bn * Hn, 32, 0, stream>>>(ckv, cn2, vbf, valbf);

  k_pv<<<dim3(Tn / 16, Bn * Hn), 32, 0, stream>>>(attn, stats, qn2, cqv, cn2, vT, cpT,
                                                  valbf);

  k_gemm_bt<<<gP, tP, 0, stream>>>(valbf, wob, bo, nullptr, out, Sn, SE, SE);
}
